// HGTConv_25881472925720
// MI455X (gfx1250) — compile-verified, matrix operand order checked by reading, not run on hardware
//
#include <hip/hip_runtime.h>
#include <cstdint>

// ---------------------------------------------------------------------------
// HGTConv collapsed form (softmax-mean over heads == 1/8 exactly):
//   S_dst   = sum_{e->dst} v_src[e]                  (f32 atomics into d_out)
//   normed  = 0.125*((S/max(cnt,1))@Wm) + 0.125*[cnt>0]*bm
//   out     = relu(normed @ Wout + bout + x)         (fused via LDS tile)
// All GEMMs: v_wmma_f32_16x16x32_bf16, bf16 inputs / f32 accumulate.
// ---------------------------------------------------------------------------

typedef __attribute__((ext_vector_type(16))) __bf16 v16bf;
typedef __attribute__((ext_vector_type(8)))  float  v8f;

union Frag16 {            // 8 VGPRs: 16 bf16 elements (A or B fragment)
  v16bf     v;
  uint16_t  u[16];
  uint4     q[2];
};

__device__ __forceinline__ uint16_t f2b(float f) {  // f32 -> bf16 (RNE)
  union { float f; uint32_t u; } c; c.f = f;
  uint32_t r = c.u + 0x7FFFu + ((c.u >> 16) & 1u);
  return (uint16_t)(r >> 16);
}
__device__ __forceinline__ float b2f(uint16_t h) {
  union { uint32_t u; float f; } c; c.u = ((uint32_t)h) << 16;
  return c.f;
}

#define WMMA_BF16(A, B, C) \
  __builtin_amdgcn_wmma_f32_16x16x32_bf16(false, (A), false, (B), (short)0, (C), false, false)

// ---------------------------------------------------------------- zero (f32x4)
__global__ void zero4_kernel(float4* __restrict__ p, long long n4) {
  long long i = (long long)blockIdx.x * blockDim.x + threadIdx.x;
  long long stride = (long long)gridDim.x * blockDim.x;
  float4 z = {0.f, 0.f, 0.f, 0.f};
  for (; i < n4; i += stride) p[i] = z;
}

// -------------------------------------------- convert W (256x256) -> bf16, T
__global__ void cvt_transpose_kernel(const float* __restrict__ W,
                                     uint16_t* __restrict__ BT) {
  int k = blockIdx.x;    // row of W
  int n = threadIdx.x;   // col of W
  BT[n * 256 + k] = f2b(W[k * 256 + n]);   // BT[n][k]
}

// --------------------------------------------------- O(bf16) = X @ W^T? no:
// O[M,256] = X[M,256] @ W[256,256] + bias ; W given transposed bf16 (BT[n][k])
__global__ __launch_bounds__(256) void gemm_bias_bf16_kernel(
    const float* __restrict__ X, const uint16_t* __restrict__ BT,
    const float* __restrict__ bias, uint16_t* __restrict__ O, int M) {
  const int wave = threadIdx.x >> 5;
  const int lane = threadIdx.x & 31;
  const int hf   = lane >> 4;          // lane half (0/1)
  const int l15  = lane & 15;
  const int mbase = blockIdx.x * 128 + wave * 16;

  // --- A fragments for all K (row = l15, K split per lane half) ---
  int arow = mbase + l15; if (arow > M - 1) arow = M - 1;
  const float* xr = X + (size_t)arow * 256;
  Frag16 af[8];
#pragma unroll
  for (int kc = 0; kc < 8; ++kc) {
    const int k0 = kc * 32 + hf * 8;
    float4 f0 = *(const float4*)(xr + k0);
    float4 f1 = *(const float4*)(xr + k0 + 4);
    float4 f2 = *(const float4*)(xr + k0 + 16);
    float4 f3 = *(const float4*)(xr + k0 + 20);
    af[kc].u[0]  = f2b(f0.x); af[kc].u[1]  = f2b(f0.y);
    af[kc].u[2]  = f2b(f0.z); af[kc].u[3]  = f2b(f0.w);
    af[kc].u[4]  = f2b(f1.x); af[kc].u[5]  = f2b(f1.y);
    af[kc].u[6]  = f2b(f1.z); af[kc].u[7]  = f2b(f1.w);
    af[kc].u[8]  = f2b(f2.x); af[kc].u[9]  = f2b(f2.y);
    af[kc].u[10] = f2b(f2.z); af[kc].u[11] = f2b(f2.w);
    af[kc].u[12] = f2b(f3.x); af[kc].u[13] = f2b(f3.y);
    af[kc].u[14] = f2b(f3.z); af[kc].u[15] = f2b(f3.w);
  }

#pragma unroll 1
  for (int nc = 0; nc < 4; ++nc) {           // 64 output cols per chunk
    v8f acc[4] = {};
#pragma unroll
    for (int nt = 0; nt < 4; ++nt) {
      const uint16_t* bp = BT + (size_t)(nc * 64 + nt * 16 + l15) * 256;
#pragma unroll
      for (int kc = 0; kc < 8; ++kc) {
        Frag16 fb;
        fb.q[0] = *(const uint4*)(bp + kc * 32 + hf * 8);
        fb.q[1] = *(const uint4*)(bp + kc * 32 + 16 + hf * 8);
        acc[nt] = WMMA_BF16(af[kc].v, fb.v, acc[nt]);
      }
    }
#pragma unroll
    for (int nt = 0; nt < 4; ++nt) {
      const int col = nc * 64 + nt * 16 + l15;
      const float bb = bias[col];
#pragma unroll
      for (int i = 0; i < 8; ++i) {
        int row = mbase + hf * 8 + i;       // C/D: M = vgpr + 8*half
        if (row < M) O[(size_t)row * 256 + col] = f2b(acc[nt][i] + bb);
      }
    }
  }
}

// -------------------------------- scatter: S[dst] += v[src]; cnt[dst] += 1
__global__ __launch_bounds__(256) void scatter_rows_kernel(
    const uint16_t* __restrict__ V, const int* __restrict__ src,
    const int* __restrict__ dst, float* __restrict__ S,
    float* __restrict__ cnt, int E) {
  int e = blockIdx.x * 8 + (threadIdx.x >> 5);   // one wave per edge
  if (e >= E) return;
  const int lane = threadIdx.x & 31;
  const uint16_t* vr = V + (size_t)src[e] * 256;
  float* srow = S + (size_t)dst[e] * 256;
  uint4 p = *(const uint4*)(vr + lane * 8);      // 8 bf16 per lane
  const uint32_t w[4] = {p.x, p.y, p.z, p.w};
#pragma unroll
  for (int j = 0; j < 4; ++j) {
    unsafeAtomicAdd(&srow[lane * 8 + 2 * j],     b2f((uint16_t)(w[j] & 0xFFFFu)));
    unsafeAtomicAdd(&srow[lane * 8 + 2 * j + 1], b2f((uint16_t)(w[j] >> 16)));
  }
  if (lane == 0) unsafeAtomicAdd(cnt + dst[e], 1.0f);
}

// ------------- fused finish: normed = 0.125*((S/mx)@Wm) + 0.125*[cnt>0]*bm
//               out = relu(normed @ Wout + bout + x)   (in-place on S/OUT)
__global__ __launch_bounds__(256) void finish_fused_kernel(
    float* __restrict__ SOUT, const float* __restrict__ cnt,
    const float* __restrict__ X,
    const uint16_t* __restrict__ WmT, const float* __restrict__ bm,
    const uint16_t* __restrict__ WoT, const float* __restrict__ bo, int M) {
  __shared__ uint16_t tile[128 * 256];           // 64KB bf16 intermediate
  const int wave = threadIdx.x >> 5;
  const int lane = threadIdx.x & 31;
  const int hf   = lane >> 4;
  const int l15  = lane & 15;
  const int mbase = blockIdx.x * 128 + wave * 16;

  // --- load A = (0.125/max(cnt,1)) * S, as bf16 fragments ---
  int arow = mbase + l15; if (arow > M - 1) arow = M - 1;
  const float ascale = 0.125f / fmaxf(cnt[arow], 1.0f);
  const float* sr = SOUT + (size_t)arow * 256;
  Frag16 af[8];
#pragma unroll
  for (int kc = 0; kc < 8; ++kc) {
    const int k0 = kc * 32 + hf * 8;
    float4 f0 = *(const float4*)(sr + k0);
    float4 f1 = *(const float4*)(sr + k0 + 4);
    float4 f2 = *(const float4*)(sr + k0 + 16);
    float4 f3 = *(const float4*)(sr + k0 + 20);
    af[kc].u[0]  = f2b(f0.x * ascale); af[kc].u[1]  = f2b(f0.y * ascale);
    af[kc].u[2]  = f2b(f0.z * ascale); af[kc].u[3]  = f2b(f0.w * ascale);
    af[kc].u[4]  = f2b(f1.x * ascale); af[kc].u[5]  = f2b(f1.y * ascale);
    af[kc].u[6]  = f2b(f1.z * ascale); af[kc].u[7]  = f2b(f1.w * ascale);
    af[kc].u[8]  = f2b(f2.x * ascale); af[kc].u[9]  = f2b(f2.y * ascale);
    af[kc].u[10] = f2b(f2.z * ascale); af[kc].u[11] = f2b(f2.w * ascale);
    af[kc].u[12] = f2b(f3.x * ascale); af[kc].u[13] = f2b(f3.y * ascale);
    af[kc].u[14] = f2b(f3.z * ascale); af[kc].u[15] = f2b(f3.w * ascale);
  }

  // bm scale per output row of this lane's C tiles (cnt>0 ? 0.125 : 0)
  float bsc[8];
#pragma unroll
  for (int i = 0; i < 8; ++i) {
    int r = mbase + hf * 8 + i;
    float c = (r < M) ? cnt[r] : 0.f;
    bsc[i] = (c > 0.f) ? 0.125f : 0.f;
  }

  // --- GEMM1: normed -> LDS tile (bf16) ---
#pragma unroll 1
  for (int nc = 0; nc < 4; ++nc) {
    v8f acc[4] = {};
#pragma unroll
    for (int nt = 0; nt < 4; ++nt) {
      const uint16_t* bp = WmT + (size_t)(nc * 64 + nt * 16 + l15) * 256;
#pragma unroll
      for (int kc = 0; kc < 8; ++kc) {
        Frag16 fb;
        fb.q[0] = *(const uint4*)(bp + kc * 32 + hf * 8);
        fb.q[1] = *(const uint4*)(bp + kc * 32 + 16 + hf * 8);
        acc[nt] = WMMA_BF16(af[kc].v, fb.v, acc[nt]);
      }
    }
#pragma unroll
    for (int nt = 0; nt < 4; ++nt) {
      const int col = nc * 64 + nt * 16 + l15;
      const float bmc = bm[col];
#pragma unroll
      for (int i = 0; i < 8; ++i) {
        const int rl = wave * 16 + hf * 8 + i;
        tile[rl * 256 + col] = f2b(acc[nt][i] + bsc[i] * bmc);
      }
    }
  }
  __syncthreads();

  // --- GEMM2: A = normed (LDS), B = Wout^T; epilogue relu(.. + bout + x) ---
  const uint16_t* tr = tile + (size_t)(wave * 16 + l15) * 256;
#pragma unroll
  for (int kc = 0; kc < 8; ++kc) {
    af[kc].q[0] = *(const uint4*)(tr + kc * 32 + hf * 8);
    af[kc].q[1] = *(const uint4*)(tr + kc * 32 + 16 + hf * 8);
  }
#pragma unroll 1
  for (int nc = 0; nc < 4; ++nc) {
    v8f acc[4] = {};
#pragma unroll
    for (int nt = 0; nt < 4; ++nt) {
      const uint16_t* bp = WoT + (size_t)(nc * 64 + nt * 16 + l15) * 256;
#pragma unroll
      for (int kc = 0; kc < 8; ++kc) {
        Frag16 fb;
        fb.q[0] = *(const uint4*)(bp + kc * 32 + hf * 8);
        fb.q[1] = *(const uint4*)(bp + kc * 32 + 16 + hf * 8);
        acc[nt] = WMMA_BF16(af[kc].v, fb.v, acc[nt]);
      }
    }
#pragma unroll
    for (int nt = 0; nt < 4; ++nt) {
      const int col = nc * 64 + nt * 16 + l15;
      const float boc = bo[col];
#pragma unroll
      for (int i = 0; i < 8; ++i) {
        int row = mbase + hf * 8 + i;
        if (row < M) {
          size_t off = (size_t)row * 256 + col;
          float v = acc[nt][i] + boc + X[off];
          SOUT[off] = fmaxf(v, 0.f);
        }
      }
    }
  }
}

// ---------------------------------------------------------------------------
extern "C" void kernel_launch(void* const* d_in, const int* in_sizes, int n_in,
                              void* d_out, int out_size, void* d_ws, size_t ws_size,
                              hipStream_t stream) {
  (void)in_sizes; (void)n_in; (void)out_size; (void)ws_size;
  const int NU = 100000, NG = 50000, E = 200000;

  const float* x_user    = (const float*)d_in[0];
  const float* x_game    = (const float*)d_in[1];
  const float* Wv_user   = (const float*)d_in[6];
  const float* bv_user   = (const float*)d_in[7];
  const float* Wout_user = (const float*)d_in[8];
  const float* bout_user = (const float*)d_in[9];
  const float* Wv_game   = (const float*)d_in[14];
  const float* bv_game   = (const float*)d_in[15];
  const float* Wout_game = (const float*)d_in[16];
  const float* bout_game = (const float*)d_in[17];
  const float* Wm_played = (const float*)d_in[20];
  const float* bm_played = (const float*)d_in[21];
  const float* Wm_rev    = (const float*)d_in[24];
  const float* bm_rev    = (const float*)d_in[25];
  const int* ei_p_src = (const int*)d_in[26];
  const int* ei_p_dst = (const int*)d_in[27];
  const int* ei_r_src = (const int*)d_in[28];
  const int* ei_r_dst = (const int*)d_in[29];
  float* out = (float*)d_out;

  // --- workspace layout (bf16 weights^T, bf16 v arrays, counts) ---
  uint16_t* WvU_T = (uint16_t*)d_ws;              // each 256*256 = 65536 elts
  uint16_t* WvG_T = WvU_T + 65536;
  uint16_t* WmP_T = WvG_T + 65536;
  uint16_t* WmR_T = WmP_T + 65536;
  uint16_t* WoU_T = WmR_T + 65536;
  uint16_t* WoG_T = WoU_T + 65536;
  uint16_t* v_u   = WoG_T + 65536;                // NU*256 bf16
  uint16_t* v_g   = v_u + (size_t)NU * 256;       // NG*256 bf16
  float*    cnt_u = (float*)(v_g + (size_t)NG * 256);
  float*    cnt_g = cnt_u + NU;
  // S (agg accumulators) live inside d_out
  float* S_u = out;                               // rows [0, NU)
  float* S_g = out + (size_t)NU * 256;            // rows [NU, NU+NG)

  // 1) zero agg (all of d_out) and counts
  zero4_kernel<<<2048, 256, 0, stream>>>((float4*)out,
                                         (long long)(NU + NG) * 256 / 4);
  zero4_kernel<<<64, 256, 0, stream>>>((float4*)cnt_u, (long long)(NU + NG) / 4);

  // 2) weights -> bf16, transposed
  cvt_transpose_kernel<<<256, 256, 0, stream>>>(Wv_user,   WvU_T);
  cvt_transpose_kernel<<<256, 256, 0, stream>>>(Wv_game,   WvG_T);
  cvt_transpose_kernel<<<256, 256, 0, stream>>>(Wm_played, WmP_T);
  cvt_transpose_kernel<<<256, 256, 0, stream>>>(Wm_rev,    WmR_T);
  cvt_transpose_kernel<<<256, 256, 0, stream>>>(Wout_user, WoU_T);
  cvt_transpose_kernel<<<256, 256, 0, stream>>>(Wout_game, WoG_T);

  // 3) v projections (bf16 out)
  gemm_bias_bf16_kernel<<<(NU + 127) / 128, 256, 0, stream>>>(
      x_user, WvU_T, bv_user, v_u, NU);
  gemm_bias_bf16_kernel<<<(NG + 127) / 128, 256, 0, stream>>>(
      x_game, WvG_T, bv_game, v_g, NG);

  // 4) scatter raw v rows + degree counts (L2-resident f32 atomics)
  scatter_rows_kernel<<<(E + 7) / 8, 256, 0, stream>>>(
      v_u, ei_p_src, ei_p_dst, S_g, cnt_g, E);   // user --played--> game
  scatter_rows_kernel<<<(E + 7) / 8, 256, 0, stream>>>(
      v_g, ei_r_src, ei_r_dst, S_u, cnt_u, E);   // game --rev--> user

  // 5) fused finish: (@Wm via LDS) then (@Wout + x, relu), in-place on d_out
  finish_fused_kernel<<<(NU + 127) / 128, 256, 0, stream>>>(
      S_u, cnt_u, x_user, WmR_T, bm_rev, WoU_T, bout_user, NU);
  finish_fused_kernel<<<(NG + 127) / 128, 256, 0, stream>>>(
      S_g, cnt_g, x_game, WmP_T, bm_played, WoG_T, bout_game, NG);
}
